// RelativePositionSelfAttention_21363167330617
// MI455X (gfx1250) — matrix operand order repaired and checked by reading, not compiled, not run on hardware
//
#include <hip/hip_runtime.h>
#include <stdint.h>

typedef _Float16 v16h __attribute__((ext_vector_type(16)));
typedef _Float16 v8h  __attribute__((ext_vector_type(8)));
typedef _Float16 v4h  __attribute__((ext_vector_type(4)));
typedef float    v8f  __attribute__((ext_vector_type(8)));

#define S_LEN 2048
#define EMB   1024
#define NH    16
#define HD    64
#define BATCH 4
#define MRD   32

// ---------------- WMMA fragment loaders (wave32, 16x16x32 f16) ----------------
// A-fragment 16x32 (MxK): row = lane&15 ; kb = 8*(lane>>4)
//   slots 0..7  -> K = kb..kb+7 ; slots 8..15 -> K = 16+kb..16+kb+7
__device__ __forceinline__ v16h load_a_f16(const _Float16* __restrict__ row, int lane) {
  const int kb = (lane >> 4) * 8;
  v8h lo = *(const v8h*)(row + kb);
  v8h hi = *(const v8h*)(row + 16 + kb);
  v16h r;
#pragma unroll
  for (int i = 0; i < 8; ++i) { r[i] = lo[i]; r[8 + i] = hi[i]; }
  return r;
}
// B-fragment 32x16 (KxN): col = lane&15 (baked into `col` pointer by caller)
//   slots 0..7  -> K = kb..kb+7 ; slots 8..15 -> K = 16+kb..16+kb+7 (kb = 8*(lane>>4))
__device__ __forceinline__ v16h load_b_f16(const _Float16* __restrict__ col, int lane) {
  const int kb = (lane >> 4) * 8;
  v8h lo = *(const v8h*)(col + kb);
  v8h hi = *(const v8h*)(col + 16 + kb);
  v16h r;
#pragma unroll
  for (int i = 0; i < 8; ++i) { r[i] = lo[i]; r[8 + i] = hi[i]; }
  return r;
}

__device__ __forceinline__ float redmax16(float v) {
  v = fmaxf(v, __shfl_xor(v, 1, 32));
  v = fmaxf(v, __shfl_xor(v, 2, 32));
  v = fmaxf(v, __shfl_xor(v, 4, 32));
  v = fmaxf(v, __shfl_xor(v, 8, 32));
  return v;
}
__device__ __forceinline__ float redsum16(float v) {
  v += __shfl_xor(v, 1, 32);
  v += __shfl_xor(v, 2, 32);
  v += __shfl_xor(v, 4, 32);
  v += __shfl_xor(v, 8, 32);
  return v;
}

// ---------------- kernel 0: f32 -> f16 pre-conversion (one pass) ----------------
__global__ __launch_bounds__(256) void cvt_f16_kernel(const float* __restrict__ in,
                                                      _Float16* __restrict__ outh, int n) {
  int i = (blockIdx.x * 256 + threadIdx.x) * 4;
  if (i + 3 < n) {
    const float4 f = *(const float4*)(in + i);
    v4h h; h[0] = (_Float16)f.x; h[1] = (_Float16)f.y; h[2] = (_Float16)f.z; h[3] = (_Float16)f.w;
    *(v4h*)(outh + i) = h;
  }
}

// ---------------- kernel 1: head-averaged relative-position bias table ----------------
__global__ void rel_mask_kernel(const float* __restrict__ rel_bias, float* __restrict__ mask_avg) {
  int t = threadIdx.x;
  if (t < 2 * MRD + 1) {
    float s = 0.f;
#pragma unroll
    for (int h = 0; h < NH; ++h) s += rel_bias[t * NH + h];
    mask_avg[t] = s * (1.0f / NH);
  }
}

// ---------------- kernel 2: fused QKV projection  qkv = x @ W^T + b ----------------
// Grid (3E/128, M/256), 256 threads = 8 waves (4 along M x 2 along N);
// wave owns a 64x64 tile (4x4 WMMA accumulators). Inputs pre-converted to f16.
// Writes Q,K as f16 [b,h,s,d]; V transposed as f16 [b,h,d,s].
__global__ __launch_bounds__(256) void qkv_proj_kernel(
    const _Float16* __restrict__ xh, const _Float16* __restrict__ wh,
    const float* __restrict__ bias,
    _Float16* __restrict__ Qb, _Float16* __restrict__ Kb, _Float16* __restrict__ Vtb)
{
  const int tid = threadIdx.x, lane = tid & 31, wid = tid >> 5;
  const int wm = wid & 3, wn = wid >> 2;
  const int m0 = blockIdx.y * 256 + wm * 64;
  const int n0 = blockIdx.x * 128 + wn * 64;
  const int mrow = lane & 15, hi = lane >> 4, nn = lane & 15;

  v8f acc[4][4] = {};
  for (int k0 = 0; k0 < EMB; k0 += 32) {
    v16h a[4], b[4];
#pragma unroll
    for (int t = 0; t < 4; ++t) {
      const _Float16* ra = xh + (size_t)(m0 + t * 16 + mrow) * EMB + k0;
      const _Float16* rb = wh + (size_t)(n0 + t * 16 + nn)   * EMB + k0;
      a[t] = load_a_f16(ra, lane);
      b[t] = load_b_f16(rb, lane);
      if (t == 0 && k0 + 32 < EMB) {           // prefetch next K-tile
        __builtin_prefetch(ra + 32, 0, 3);
        __builtin_prefetch(rb + 32, 0, 3);
      }
    }
#pragma unroll
    for (int tm = 0; tm < 4; ++tm)
#pragma unroll
      for (int tn = 0; tn < 4; ++tn)
        acc[tm][tn] = __builtin_amdgcn_wmma_f32_16x16x32_f16(
            false, a[tm], false, b[tn], (short)0, acc[tm][tn], false, false);
  }

#pragma unroll
  for (int tm = 0; tm < 4; ++tm)
#pragma unroll
    for (int tn = 0; tn < 4; ++tn)
#pragma unroll
      for (int r = 0; r < 8; ++r) {
        const int M = m0 + tm * 16 + r + 8 * hi;     // row in [0, B*S)
        const int N = n0 + tn * 16 + nn;             // col in [0, 3E)
        const float v = acc[tm][tn][r] + bias[N];
        const int which = N >> 10;                   // 0:q 1:k 2:v
        const int rem = N & (EMB - 1);
        const int h = rem >> 6, d = rem & (HD - 1);
        const int bi = M >> 11, s = M & (S_LEN - 1);
        const _Float16 hv = (_Float16)v;
        const size_t bh = (size_t)bi * NH + h;
        if (which == 0)      Qb [(bh * S_LEN + s) * HD + d] = hv;
        else if (which == 1) Kb [(bh * S_LEN + s) * HD + d] = hv;
        else                 Vtb[(bh * HD + d) * S_LEN + s] = hv;  // transposed
      }
}

// ---------------- kernel 3: flash attention (online softmax) ----------------
// Grid (S/128, B*H); wave handles 16 query rows, streams keys 32 at a time.
__global__ __launch_bounds__(256) void flash_attn_kernel(
    const _Float16* __restrict__ Qb, const _Float16* __restrict__ Kb,
    const _Float16* __restrict__ Vtb, const float* __restrict__ mask_avg,
    _Float16* __restrict__ ctx)
{
  __shared__ float mlds[2 * MRD + 1];
  __shared__ __align__(16) _Float16 Plds[8][16 * 32];   // per-wave 16x32 P tile

  const int tid = threadIdx.x, lane = tid & 31, wv = tid >> 5;

  // async memory->LDS copy of the 65-entry mask table (ASYNCcnt-tracked)
  if (tid < 2 * MRD + 1) {
    uint32_t lds_off = (uint32_t)(uintptr_t)&mlds[tid];
    uint64_t gaddr   = (uint64_t)(uintptr_t)(mask_avg + tid);
    asm volatile("global_load_async_to_lds_b32 %0, %1, off"
                 :: "v"(lds_off), "v"(gaddr) : "memory");
  }
  asm volatile("s_wait_asynccnt 0x0" ::: "memory");
  __syncthreads();

  const int bh = blockIdx.y;
  const int bi = bh >> 4, h = bh & (NH - 1);
  const size_t hoff = (size_t)bh * S_LEN * HD;          // same extent for Q,K,Vt
  const int q0 = blockIdx.x * 128 + wv * 16;
  const int mrow = lane & 15, hi = lane >> 4, nn = lane & 15;

  // Q fragments: 16 rows x 64 d  -> two 16x32 A-fragments
  v16h qa[2];
#pragma unroll
  for (int c = 0; c < 2; ++c)
    qa[c] = load_a_f16(Qb + hoff + (size_t)(q0 + mrow) * HD + c * 32, lane);

  v8f  O[4] = {};                                       // 16 x 64 accumulator
  float m[8], l[8];
#pragma unroll
  for (int r = 0; r < 8; ++r) { m[r] = -3.0e38f; l[r] = 0.f; }

  for (int k0 = 0; k0 < S_LEN; k0 += 32) {
    // ---- scores: two 16x16 tiles (keys k0..k0+15, k0+16..k0+31) ----
    v8f s0 = {}, s1 = {};
#pragma unroll
    for (int c = 0; c < 2; ++c) {
      v16h kf0 = load_b_f16(Kb + hoff + (size_t)(k0 + nn)      * HD + c * 32, lane);
      v16h kf1 = load_b_f16(Kb + hoff + (size_t)(k0 + 16 + nn) * HD + c * 32, lane);
      s0 = __builtin_amdgcn_wmma_f32_16x16x32_f16(false, qa[c], false, kf0, (short)0, s0, false, false);
      s1 = __builtin_amdgcn_wmma_f32_16x16x32_f16(false, qa[c], false, kf1, (short)0, s1, false, false);
    }

    // ---- scale + relative-position mask + online softmax ----
    float p0[8], p1[8], alpha[8];
#pragma unroll
    for (int r = 0; r < 8; ++r) {
      const int iq = q0 + r + 8 * hi;
      int d0 = iq - (k0 + nn);
      int d1 = iq - (k0 + 16 + nn);
      d0 = d0 < -MRD ? -MRD : (d0 > MRD ? MRD : d0);
      d1 = d1 < -MRD ? -MRD : (d1 > MRD ? MRD : d1);
      const float v0 = s0[r] * 0.125f + mlds[d0 + MRD];
      const float v1 = s1[r] * 0.125f + mlds[d1 + MRD];
      const float mn = fmaxf(m[r], redmax16(fmaxf(v0, v1)));
      alpha[r] = __expf(m[r] - mn);
      p0[r] = __expf(v0 - mn);
      p1[r] = __expf(v1 - mn);
      l[r] = l[r] * alpha[r] + redsum16(p0[r] + p1[r]);
      m[r] = mn;
    }
#pragma unroll
    for (int n = 0; n < 4; ++n)
#pragma unroll
      for (int r = 0; r < 8; ++r) O[n][r] *= alpha[r];

    // ---- transpose P (C-layout -> A-layout) through wave-private LDS ----
#pragma unroll
    for (int r = 0; r < 8; ++r) {
      Plds[wv][(r + 8 * hi) * 32 + nn]      = (_Float16)p0[r];
      Plds[wv][(r + 8 * hi) * 32 + 16 + nn] = (_Float16)p1[r];
    }
    __syncthreads();   // uniform across all waves every iteration

    const v16h pa = load_a_f16(&Plds[wv][mrow * 32], lane);

    // ---- O += P @ V : V^T is [d, s], columns contiguous ----
#pragma unroll
    for (int n = 0; n < 4; ++n) {
      v16h vf = load_b_f16(Vtb + hoff + (size_t)(n * 16 + nn) * S_LEN + k0, lane);
      O[n] = __builtin_amdgcn_wmma_f32_16x16x32_f16(false, pa, false, vf, (short)0, O[n], false, false);
    }
  }

  // ---- normalize and write ctx as f16 [b, s, h*d] ----
  float inv[8];
#pragma unroll
  for (int r = 0; r < 8; ++r) inv[r] = 1.0f / l[r];
#pragma unroll
  for (int n = 0; n < 4; ++n)
#pragma unroll
    for (int r = 0; r < 8; ++r) {
      const int iq = q0 + r + 8 * hi;
      const int d = n * 16 + nn;
      ctx[((size_t)bi * S_LEN + iq) * EMB + h * HD + d] = (_Float16)(O[n][r] * inv[r]);
    }
}

// ---------------- kernel 4: output projection  out = ctx @ W^T + b (f32) ----------------
// Same 64x64-per-wave tiling; A = ctx (f16), B = pre-converted out_w (f16).
__global__ __launch_bounds__(256) void out_proj_kernel(
    const _Float16* __restrict__ ctx, const _Float16* __restrict__ wh,
    const float* __restrict__ bias, float* __restrict__ out)
{
  const int tid = threadIdx.x, lane = tid & 31, wid = tid >> 5;
  const int wm = wid & 3, wn = wid >> 2;
  const int m0 = blockIdx.y * 256 + wm * 64;
  const int n0 = blockIdx.x * 128 + wn * 64;
  const int mrow = lane & 15, hi = lane >> 4, nn = lane & 15;

  v8f acc[4][4] = {};
  for (int k0 = 0; k0 < EMB; k0 += 32) {
    v16h a[4], b[4];
#pragma unroll
    for (int t = 0; t < 4; ++t) {
      const _Float16* ra = ctx + (size_t)(m0 + t * 16 + mrow) * EMB + k0;
      const _Float16* rb = wh  + (size_t)(n0 + t * 16 + nn)   * EMB + k0;
      a[t] = load_a_f16(ra, lane);
      b[t] = load_b_f16(rb, lane);
      if (t == 0 && k0 + 32 < EMB) {
        __builtin_prefetch(ra + 32, 0, 3);
        __builtin_prefetch(rb + 32, 0, 3);
      }
    }
#pragma unroll
    for (int tm = 0; tm < 4; ++tm)
#pragma unroll
      for (int tn = 0; tn < 4; ++tn)
        acc[tm][tn] = __builtin_amdgcn_wmma_f32_16x16x32_f16(
            false, a[tm], false, b[tn], (short)0, acc[tm][tn], false, false);
  }

#pragma unroll
  for (int tm = 0; tm < 4; ++tm)
#pragma unroll
    for (int tn = 0; tn < 4; ++tn)
#pragma unroll
      for (int r = 0; r < 8; ++r) {
        const int M = m0 + tm * 16 + r + 8 * hi;
        const int N = n0 + tn * 16 + nn;
        out[(size_t)M * EMB + N] = acc[tm][tn][r] + bias[N];
      }
}

// ---------------- launch ----------------
extern "C" void kernel_launch(void* const* d_in, const int* in_sizes, int n_in,
                              void* d_out, int out_size, void* d_ws, size_t ws_size,
                              hipStream_t stream) {
  const float* x         = (const float*)d_in[0];   // [4,2048,1024]
  const float* in_proj_w = (const float*)d_in[1];   // [3072,1024]
  const float* in_proj_b = (const float*)d_in[2];   // [3072]
  const float* out_w     = (const float*)d_in[3];   // [1024,1024]
  const float* out_b     = (const float*)d_in[4];   // [1024]
  const float* rel_bias  = (const float*)d_in[5];   // [65,16]
  float* out = (float*)d_out;                       // [4,2048,1024] f32

  // workspace layout
  const size_t HBUF = (size_t)BATCH * NH * S_LEN * HD;   // 8,388,608 halves
  const size_t NX   = (size_t)BATCH * S_LEN * EMB;       // 8,388,608
  const size_t NW   = (size_t)3 * EMB * EMB;             // 3,145,728
  const size_t NOW  = (size_t)EMB * EMB;                 // 1,048,576
  char* ws = (char*)d_ws;
  float*     mask_avg = (float*)ws;                      // 65 f32
  _Float16*  Qb   = (_Float16*)(ws + 512);
  _Float16*  Kb   = Qb   + HBUF;
  _Float16*  Vtb  = Kb   + HBUF;
  _Float16*  ctxb = Vtb  + HBUF;
  _Float16*  xh   = ctxb + HBUF;
  _Float16*  wh   = xh   + NX;
  _Float16*  owh  = wh   + NW;

  rel_mask_kernel<<<1, 128, 0, stream>>>(rel_bias, mask_avg);
  cvt_f16_kernel<<<(int)(NX  / 1024), 256, 0, stream>>>(x,         xh,  (int)NX);
  cvt_f16_kernel<<<(int)(NW  / 1024), 256, 0, stream>>>(in_proj_w, wh,  (int)NW);
  cvt_f16_kernel<<<(int)(NOW / 1024), 256, 0, stream>>>(out_w,     owh, (int)NOW);

  qkv_proj_kernel<<<dim3(3 * EMB / 128, (BATCH * S_LEN) / 256), 256, 0, stream>>>(
      xh, wh, in_proj_b, Qb, Kb, Vtb);

  flash_attn_kernel<<<dim3(S_LEN / 128, BATCH * NH), 256, 0, stream>>>(
      Qb, Kb, Vtb, mask_avg, ctxb);

  out_proj_kernel<<<dim3(EMB / 128, (BATCH * S_LEN) / 256), 256, 0, stream>>>(
      ctxb, owh, out_b, out);
}
